// PaiNN_11089605558372
// MI455X (gfx1250) — matrix operand order repaired and checked by reading, not compiled, not run on hardware
//
#include <hip/hip_runtime.h>
#include <hip/hip_bf16.h>
#include <math.h>

// ---- problem constants (match reference) ----
#define NN   20000
#define EE   320000
#define FF   128
#define EDD  20
#define RRND 3
#define GGR  256
#define PIF  3.14159265358979f
#define CUTV 5.0f
#define NF   ((size_t)NN * FF)   // 2,560,000

typedef __attribute__((ext_vector_type(16))) _Float16 v16h;
typedef __attribute__((ext_vector_type(8)))  _Float16 v8h;
typedef __attribute__((ext_vector_type(8)))  float    v8f;

// =======================================================================
// WMMA GEMM, K templated (fully unrolled), 16x32 output per wave.
// A: MxK f16 row-major.  Bsw: pre-swizzled fragment-order f16:
//   element ((nt*(K/32)+kc)*32 + lane)*16 + i  ==  B[kc*32 + (lane>>4)*16 + i][nt*16 + (lane&15)]
// so each lane's B fragment is one contiguous 32-byte load.
// A fragment (contiguous k order, same as Bsw): lane m=l&15, half=l>>4,
//   halves 0..7  = A[m][k0 + half*16 + i]      (contiguous 16B)
//   halves 8..15 = A[m][k0 + half*16 + 8 + i]  (contiguous 16B)
// =======================================================================
template <int K>
__global__ __launch_bounds__(256) void gemm_wmma(
    const _Float16* __restrict__ A, const _Float16* __restrict__ Bsw,
    const float* __restrict__ bias, float* __restrict__ outF,
    _Float16* __restrict__ outH, int M, int Nc, int doSilu)
{
    constexpr int KC = K / 32;
    int lane = threadIdx.x & 31;
    int wave = threadIdx.x >> 5;
    int npairs = Nc >> 5;                       // pairs of 16-wide N tiles
    int tile = blockIdx.x * 8 + wave;
    if (tile >= (M >> 4) * npairs) return;      // never taken with our dims
    int mt = tile / npairs, np = tile - mt * npairs;
    int nt0 = np * 2, nt1 = nt0 + 1;
    int half = lane >> 4;
    int lan  = lane & 15;
    int mrow = mt * 16 + lan;

    v8f c0 = {}, c1 = {};
    const _Float16* arow = A + (size_t)mrow * K;
    const _Float16* b0 = Bsw + (size_t)nt0 * KC * 512;   // 32 lanes * 16 halfs per block
    const _Float16* b1 = Bsw + (size_t)nt1 * KC * 512;
#pragma unroll
    for (int kc = 0; kc < KC; ++kc) {
        v8h alo = *(const v8h*)(arow + kc * 32 + half * 16);
        v8h ahi = *(const v8h*)(arow + kc * 32 + half * 16 + 8);
        v16h a;
#pragma unroll
        for (int i = 0; i < 8; ++i) { a[i] = alo[i]; a[i + 8] = ahi[i]; }
        v16h bb0 = *(const v16h*)(b0 + (size_t)(kc * 32 + lane) * 16);
        v16h bb1 = *(const v16h*)(b1 + (size_t)(kc * 32 + lane) * 16);
        c0 = __builtin_amdgcn_wmma_f32_16x16x32_f16(false, a, false, bb0, (short)0, c0, false, false);
        c1 = __builtin_amdgcn_wmma_f32_16x16x32_f16(false, a, false, bb1, (short)0, c1, false, false);
    }
    float bv0 = bias ? bias[nt0 * 16 + lan] : 0.0f;
    float bv1 = bias ? bias[nt1 * 16 + lan] : 0.0f;
#pragma unroll
    for (int v = 0; v < 8; ++v) {
        int row = mt * 16 + v + 8 * half;
        float x0 = c0[v] + bv0;
        float x1 = c1[v] + bv1;
        if (doSilu) {
            x0 = x0 / (1.0f + __expf(-x0));
            x1 = x1 / (1.0f + __expf(-x1));
        }
        size_t base = (size_t)row * Nc;
        if (outF) { outF[base + nt0 * 16 + lan] = x0; outF[base + nt1 * 16 + lan] = x1; }
        if (outH) { outH[base + nt0 * 16 + lan] = (_Float16)x0; outH[base + nt1 * 16 + lan] = (_Float16)x1; }
    }
}

// ---- weight -> WMMA-fragment-order swizzle (f32 -> f16), optional transpose ----
// trans=0: logical B[k][n] = src[k*Nc + n]   (plain K x Nc weight)
// trans=1: logical B[k][n] = src[n*K + k]    (B = src^T, src is Nc x K)
__global__ void swizzleB(const float* __restrict__ src, _Float16* __restrict__ dst,
                         int K, int Nc, int trans, int total) {
    int o = blockIdx.x * blockDim.x + threadIdx.x;
    if (o >= total) return;
    int i = o & 15;
    int lane = (o >> 4) & 31;
    int blkIdx = o >> 9;
    int KC = K >> 5;
    int kc = blkIdx % KC;
    int nt = blkIdx / KC;
    int n = lane & 15, half = lane >> 4;
    int k = kc * 32 + half * 16 + i;
    int col = nt * 16 + n;
    float v = trans ? src[(size_t)col * K + k] : src[(size_t)k * Nc + col];
    dst[o] = (_Float16)v;
}

__global__ void zero_f32(float* p, int n) {
    int i = blockIdx.x * blockDim.x + threadIdx.x;
    if (i < n) p[i] = 0.0f;
}

__global__ void edge_geom(const float* __restrict__ pos, const int* __restrict__ ei,
                          float* __restrict__ rhat, float* __restrict__ cutb,
                          float* __restrict__ rbf) {
    int e = blockIdx.x * blockDim.x + threadIdx.x;
    if (e >= EE) return;
    int r = ei[e], c = ei[EE + e];
    float dx = pos[c * 3 + 0] - pos[r * 3 + 0];
    float dy = pos[c * 3 + 1] - pos[r * 3 + 1];
    float dz = pos[c * 3 + 2] - pos[r * 3 + 2];
    float nrm = sqrtf(dx * dx + dy * dy + dz * dz);
    float inv = 1.0f / nrm;
    rhat[e * 3 + 0] = dx * inv; rhat[e * 3 + 1] = dy * inv; rhat[e * 3 + 2] = dz * inv;
    cutb[e] = (nrm < CUTV) ? 0.5f * (__cosf(PIF * nrm / CUTV) + 1.0f) : 0.0f;
#pragma unroll
    for (int k = 0; k < EDD; ++k)
        rbf[e * EDD + k] = __sinf(nrm * (float)(k + 1) * PIF / CUTV) * inv;
}

__global__ void init_state(const int* __restrict__ z, const float* __restrict__ emb_w,
                           const float* __restrict__ emb_b, float* __restrict__ st,
                           _Float16* __restrict__ st16, float* __restrict__ sv,
                           _Float16* __restrict__ sv16) {
    int idx = blockIdx.x * blockDim.x + threadIdx.x;
    if (idx >= (int)NF) return;
    int n = idx >> 7, f = idx & 127;
    float s = emb_w[z[n] * FF + f] + emb_b[f];
    st[idx] = s; st16[idx] = (_Float16)s;
#pragma unroll
    for (int d = 0; d < 3; ++d) { sv[d * NF + idx] = 0.0f; sv16[d * NF + idx] = (_Float16)0.0f; }
}

// fused edge message: one block (128 threads) per edge; per-edge data staged in LDS.
// Wm = (rbf@msgW+b)*cut, comb = phi[col]*Wm, scatter deltas (read old sv, write deltas).
__global__ __launch_bounds__(128) void edge_msg(
    const int* __restrict__ ei, const float* __restrict__ rbf,
    const float* __restrict__ cutb, const float* __restrict__ rhat,
    const float* __restrict__ phi, const float* __restrict__ sv,
    const float* __restrict__ mW, const float* __restrict__ mB,
    float* __restrict__ dstate, float* __restrict__ dsv) {
    __shared__ float srbf[EDD];
    __shared__ float srh[3];
    __shared__ float scu;
    __shared__ int srow, scol;
    int e = blockIdx.x;
    int f = threadIdx.x;
    if (f < EDD) srbf[f] = rbf[(size_t)e * EDD + f];
    if (f >= 32 && f < 35) srh[f - 32] = rhat[e * 3 + (f - 32)];
    if (f == 35) scu = cutb[e];
    if (f == 36) srow = ei[e];
    if (f == 37) scol = ei[EE + e];
    __syncthreads();
    int row = srow, col = scol;
    float w0 = mB[f], w1 = mB[FF + f], w2 = mB[2 * FF + f];
#pragma unroll
    for (int k = 0; k < EDD; ++k) {
        float rv = srbf[k];
        const float* mw = mW + (size_t)k * (3 * FF);
        w0 += rv * mw[f]; w1 += rv * mw[FF + f]; w2 += rv * mw[2 * FF + f];
    }
    float cu = scu;
    const float* ph = phi + (size_t)col * (3 * FF);
    float gsv = ph[f] * w0 * cu;
    float gev = ph[FF + f] * w1 * cu;
    float sm  = ph[2 * FF + f] * w2 * cu;
    atomicAdd(&dstate[(size_t)row * FF + f], sm);
#pragma unroll
    for (int d = 0; d < 3; ++d) {
        float vm = sv[d * NF + (size_t)col * FF + f] * gsv + srh[d] * gev;
        atomicAdd(&dsv[d * NF + (size_t)row * FF + f], vm);
    }
}

__global__ void apply_delta(float* __restrict__ st, const float* __restrict__ dst,
                            float* __restrict__ sv, const float* __restrict__ dsv,
                            _Float16* __restrict__ st16, _Float16* __restrict__ sv16) {
    int idx = blockIdx.x * blockDim.x + threadIdx.x;
    if (idx >= (int)NF) return;
    float s = st[idx] + dst[idx];
    st[idx] = s; st16[idx] = (_Float16)s;
#pragma unroll
    for (int d = 0; d < 3; ++d) {
        size_t o = d * NF + idx;
        float v = sv[o] + dsv[o];
        sv[o] = v; sv16[o] = (_Float16)v;
    }
}

// cat16 = [state, norm_d(planes)] in f16 (N x 256)
__global__ void make_cat(const float* __restrict__ st, const float* __restrict__ pl,
                         _Float16* __restrict__ cat16) {
    int idx = blockIdx.x * blockDim.x + threadIdx.x;
    if (idx >= (int)NF) return;
    int n = idx >> 7, f = idx & 127;
    float a = pl[idx], b = pl[NF + idx], c = pl[2 * NF + idx];
    float vn = sqrtf(a * a + b * b + c * c);
    cat16[(size_t)n * 256 + f]       = (_Float16)st[idx];
    cat16[(size_t)n * 256 + 128 + f] = (_Float16)vn;
}

__global__ void update_a(float* __restrict__ st, float* __restrict__ sv,
                         const float* __restrict__ ud, const float* __restrict__ vd,
                         const float* __restrict__ a, _Float16* __restrict__ st16,
                         _Float16* __restrict__ sv16) {
    int idx = blockIdx.x * blockDim.x + threadIdx.x;
    if (idx >= (int)NF) return;
    int n = idx >> 7, f = idx & 127;
    float avv = a[(size_t)n * 384 + f];
    float asv = a[(size_t)n * 384 + 128 + f];
    float ass = a[(size_t)n * 384 + 256 + f];
    float dot = 0.0f;
#pragma unroll
    for (int d = 0; d < 3; ++d) {
        size_t o = d * NF + idx;
        float u = ud[o], v = vd[o];
        dot += u * v;
        float nv = sv[o] + u * avv;
        sv[o] = nv; sv16[o] = (_Float16)nv;
    }
    float s = st[idx] + dot * asv + ass;
    st[idx] = s; st16[idx] = (_Float16)s;
}

__global__ void update_geb(float* __restrict__ st, float* __restrict__ sv,
                           const float* __restrict__ Wv, const float* __restrict__ sp,
                           _Float16* __restrict__ st16, _Float16* __restrict__ sv16) {
    int idx = blockIdx.x * blockDim.x + threadIdx.x;
    if (idx >= (int)NF) return;
    int n = idx >> 7, f = idx & 127;
    float s = sp[(size_t)n * 256 + f];
    st[idx] = s; st16[idx] = (_Float16)s;
    float vh = sp[(size_t)n * 256 + 128 + f];
#pragma unroll
    for (int d = 0; d < 3; ++d) {
        size_t o = d * NF + idx;
        float nv = Wv[o] * vh;
        sv[o] = nv; sv16[o] = (_Float16)nv;
    }
}

__global__ void final_mu(const float* __restrict__ st, const float* __restrict__ sv,
                         const float* __restrict__ pos, const float* __restrict__ mu_w,
                         const float* __restrict__ q_w, const float* __restrict__ q_b,
                         const int* __restrict__ batch, float* __restrict__ out) {
    int n = blockIdx.x * blockDim.x + threadIdx.x;
    if (n >= NN) return;
    float q = q_b[0];
    for (int f = 0; f < FF; ++f) q += st[(size_t)n * FF + f] * q_w[f];
    float mu[3];
#pragma unroll
    for (int d = 0; d < 3; ++d) {
        float acc = 0.0f;
        for (int f = 0; f < FF; ++f) acc += sv[d * NF + (size_t)n * FF + f] * mu_w[f];
        mu[d] = acc + q * pos[n * 3 + d];
    }
    float mn = sqrtf(mu[0] * mu[0] + mu[1] * mu[1] + mu[2] * mu[2]);
    atomicAdd(&out[batch[n]], mn);
}

// =======================================================================
extern "C" void kernel_launch(void* const* d_in, const int* in_sizes, int n_in,
                              void* d_out, int out_size, void* d_ws, size_t ws_size,
                              hipStream_t stream) {
    const int*   z      = (const int*)d_in[0];
    const float* pos    = (const float*)d_in[1];
    const int*   ei     = (const int*)d_in[2];
    const int*   batch  = (const int*)d_in[3];
    const float* emb_w  = (const float*)d_in[4];
    const float* emb_b  = (const float*)d_in[5];
    const float* phi_w1 = (const float*)d_in[6];
    const float* phi_b1 = (const float*)d_in[7];
    const float* phi_w2 = (const float*)d_in[8];
    const float* phi_b2 = (const float*)d_in[9];
    const float* msgW_w = (const float*)d_in[10];
    const float* msgW_b = (const float*)d_in[11];
    const float* Uw     = (const float*)d_in[12];
    const float* Vw     = (const float*)d_in[13];
    const float* a_w1   = (const float*)d_in[14];
    const float* a_b1   = (const float*)d_in[15];
    const float* a_w2   = (const float*)d_in[16];
    const float* a_b2   = (const float*)d_in[17];
    const float* geb_W  = (const float*)d_in[18];
    const float* geb_w1 = (const float*)d_in[19];
    const float* geb_b1 = (const float*)d_in[20];
    const float* geb_w2 = (const float*)d_in[21];
    const float* geb_b2 = (const float*)d_in[22];
    const float* mu_w   = (const float*)d_in[23];
    const float* q_w    = (const float*)d_in[24];
    const float* q_b    = (const float*)d_in[25];
    float* out = (float*)d_out;

    // ---- workspace carve-out ----
    char* w = (char*)d_ws; size_t off = 0;
    auto alloc = [&](size_t bytes) { void* p = w + off; off = (off + bytes + 255) & ~(size_t)255; return p; };
    float*    rhat   = (float*)alloc(EE * 3 * 4);
    float*    cutb   = (float*)alloc(EE * 4);
    float*    rbf    = (float*)alloc((size_t)EE * EDD * 4);
    float*    state  = (float*)alloc(NF * 4);
    _Float16* st16   = (_Float16*)alloc(NF * 2);
    float*    sv     = (float*)alloc(3 * NF * 4);
    _Float16* sv16   = (_Float16*)alloc(3 * NF * 2);
    float*    dstate = (float*)alloc(NF * 4);
    float*    dsv    = (float*)alloc(3 * NF * 4);
    float*    buf384 = (float*)alloc((size_t)NN * 384 * 4);   // phi / a / sp
    float*    udot   = (float*)alloc(3 * NF * 4);             // also Wv in GEB
    float*    vdot   = (float*)alloc(3 * NF * 4);
    _Float16* cat16  = (_Float16*)alloc((size_t)NN * 256 * 2);
    _Float16* tmp16  = (_Float16*)alloc((size_t)NN * 256 * 2); // h16 / t16
    _Float16* wphi1  = (_Float16*)alloc((size_t)RRND * 128 * 128 * 2);
    _Float16* wphi2  = (_Float16*)alloc((size_t)RRND * 128 * 384 * 2);
    _Float16* wa1    = (_Float16*)alloc((size_t)RRND * 256 * 128 * 2);
    _Float16* wa2    = (_Float16*)alloc((size_t)RRND * 128 * 384 * 2);
    _Float16* wg1    = (_Float16*)alloc((size_t)RRND * 256 * 256 * 2);
    _Float16* wg2    = (_Float16*)alloc((size_t)RRND * 256 * 256 * 2);
    _Float16* Ut     = (_Float16*)alloc((size_t)RRND * 128 * 128 * 2);
    _Float16* Vt     = (_Float16*)alloc((size_t)RRND * 128 * 128 * 2);
    _Float16* Wt     = (_Float16*)alloc((size_t)RRND * 128 * 128 * 2);

    auto blk = [](int n) { return (n + 255) / 256; };
    auto gemm = [&](const _Float16* A, const _Float16* B, const float* bias,
                    float* oF, _Float16* oH, int M, int K, int Nc, int silu) {
        int tiles = (M / 16) * (Nc / 32);
        int blocks = (tiles + 7) / 8;
        if (K == 128)
            gemm_wmma<128><<<blocks, 256, 0, stream>>>(A, B, bias, oF, oH, M, Nc, silu);
        else
            gemm_wmma<256><<<blocks, 256, 0, stream>>>(A, B, bias, oF, oH, M, Nc, silu);
    };
    // per-r fragment-order weight swizzles (re-done every call; no caching)
    auto swz = [&](const float* src, _Float16* dst, int K, int Nc, int trans) {
        for (int r = 0; r < RRND; ++r)
            swizzleB<<<blk(K * Nc), 256, 0, stream>>>(src + (size_t)r * K * Nc,
                                                      dst + (size_t)r * K * Nc,
                                                      K, Nc, trans, K * Nc);
    };
    swz(phi_w1, wphi1, 128, 128, 0);
    swz(phi_w2, wphi2, 128, 384, 0);
    swz(a_w1,   wa1,   256, 128, 0);
    swz(a_w2,   wa2,   128, 384, 0);
    swz(geb_w1, wg1,   256, 256, 0);
    swz(geb_w2, wg2,   256, 256, 0);
    swz(Uw,     Ut,    128, 128, 1);   // B = U^T
    swz(Vw,     Vt,    128, 128, 1);
    swz(geb_W,  Wt,    128, 128, 1);

    // ---- geometry + node init ----
    edge_geom<<<blk(EE), 256, 0, stream>>>(pos, ei, rhat, cutb, rbf);
    init_state<<<blk((int)NF), 256, 0, stream>>>(z, emb_w, emb_b, state, st16, sv, sv16);

    // ---- message-passing rounds ----
    for (int r = 0; r < RRND; ++r) {
        gemm(st16, wphi1 + (size_t)r*128*128, phi_b1 + r*128, nullptr, tmp16, NN, 128, 128, 1);
        gemm(tmp16, wphi2 + (size_t)r*128*384, phi_b2 + r*384, buf384, nullptr, NN, 128, 384, 0);
        zero_f32<<<blk((int)NF), 256, 0, stream>>>(dstate, (int)NF);
        zero_f32<<<blk((int)(3*NF)), 256, 0, stream>>>(dsv, (int)(3*NF));
        edge_msg<<<EE, 128, 0, stream>>>(ei, rbf, cutb, rhat, buf384, sv,
                                         msgW_w + (size_t)r*EDD*384, msgW_b + r*384,
                                         dstate, dsv);
        apply_delta<<<blk((int)NF), 256, 0, stream>>>(state, dstate, sv, dsv, st16, sv16);
        for (int d = 0; d < 3; ++d)
            gemm(sv16 + d*NF, Ut + (size_t)r*128*128, nullptr, udot + d*NF, nullptr, NN, 128, 128, 0);
        for (int d = 0; d < 3; ++d)
            gemm(sv16 + d*NF, Vt + (size_t)r*128*128, nullptr, vdot + d*NF, nullptr, NN, 128, 128, 0);
        make_cat<<<blk((int)NF), 256, 0, stream>>>(state, vdot, cat16);
        gemm(cat16, wa1 + (size_t)r*256*128, a_b1 + r*128, nullptr, tmp16, NN, 256, 128, 1);
        gemm(tmp16, wa2 + (size_t)r*128*384, a_b2 + r*384, buf384, nullptr, NN, 128, 384, 0);
        update_a<<<blk((int)NF), 256, 0, stream>>>(state, sv, udot, vdot, buf384, st16, sv16);
    }

    // ---- gated equivariant blocks ----
    for (int r = 0; r < RRND; ++r) {
        for (int d = 0; d < 3; ++d)   // Wv in udot buffer
            gemm(sv16 + d*NF, Wt + (size_t)r*128*128, nullptr, udot + d*NF, nullptr, NN, 128, 128, 0);
        make_cat<<<blk((int)NF), 256, 0, stream>>>(state, udot, cat16);
        gemm(cat16, wg1 + (size_t)r*256*256, geb_b1 + r*256, nullptr, tmp16, NN, 256, 256, 1);
        gemm(tmp16, wg2 + (size_t)r*256*256, geb_b2 + r*256, buf384, nullptr, NN, 256, 256, 0);
        update_geb<<<blk((int)NF), 256, 0, stream>>>(state, sv, udot, buf384, st16, sv16);
    }

    // ---- readout ----
    zero_f32<<<1, 256, 0, stream>>>(out, GGR);
    final_mu<<<blk(NN), 256, 0, stream>>>(state, sv, pos, mu_w, q_w, q_b, batch, out);
}
